// Attention_34961033790132
// MI455X (gfx1250) — matrix operand order repaired and checked by reading, not compiled, not run on hardware
//
#include <hip/hip_runtime.h>

// ---------- POD vector types ----------
typedef __attribute__((ext_vector_type(16))) _Float16 v16h;
typedef __attribute__((ext_vector_type(8)))  float    v8f;
typedef __attribute__((ext_vector_type(2)))  __fp16   fp16x2;   // cvt_pkrtz return type
typedef __attribute__((ext_vector_type(4)))  _Float16 h4v;
typedef __attribute__((ext_vector_type(4)))  float    f4v;
typedef __attribute__((ext_vector_type(4)))  unsigned u4v;
typedef __attribute__((ext_vector_type(4)))  unsigned uint32x4;
typedef __attribute__((ext_vector_type(8)))  int      int32x8;
typedef __attribute__((ext_vector_type(4)))  int      int32x4;

union Frag16 { v16h v; _Float16 h[16]; u4v u4[2]; unsigned w[8]; };
union F4  { f4v v; float f[4]; };
union H2U { fp16x2 v; unsigned u; };

#define LOG2E   1.4426950408889634f
#define NEGBIG  1.25e9f                 /* INF(1e10)/sqrt(64) */
#define BIGL2   (NEGBIG * LOG2E)
#define SCL2E   (0.125f * LOG2E)        /* 1/sqrt(64) * log2(e) */

#if __has_builtin(__builtin_amdgcn_tensor_load_to_lds)
#define USE_TDM 1
#else
#define USE_TDM 0
#endif

__device__ __forceinline__ unsigned pk2(float a, float b) {
  H2U t; t.v = __builtin_amdgcn_cvt_pkrtz(a, b); return t.u;
}

#if USE_TDM
// Loop-invariant group-1 descriptor: f16 2D tile (tile_d1 rows x tile_d0 elems,
// row stride stride0 elems), packed contiguously into LDS.
__device__ __forceinline__ int32x8 tdm_g1(unsigned tile_d0, unsigned tile_d1,
                                          unsigned tensor_d0, unsigned tensor_d1,
                                          unsigned long long stride0) {
  int32x8 g1;
  g1[0] = (int)(1u << 16);                      // data_size=1 (2B), wg_mask=0
  g1[1] = (int)((tensor_d0 & 0xffffu) << 16);   // tensor_dim0[15:0]
  g1[2] = (int)((tensor_d0 >> 16) | ((tensor_d1 & 0xffffu) << 16));
  g1[3] = (int)((tensor_d1 >> 16) | (tile_d0 << 16));
  g1[4] = (int)(tile_d1 & 0xffffu);             // tile_dim1 (tile_dim2=0)
  g1[5] = (int)(unsigned)(stride0 & 0xffffffffu);
  g1[6] = (int)(unsigned)((stride0 >> 32) & 0xffffu);
  g1[7] = 0;
  return g1;
}
__device__ __forceinline__ void tdm_issue(unsigned lds_addr, unsigned long long gaddr,
                                          int32x8 g1) {
  uint32x4 g0;
  g0[0] = 1u;                                   // count=1, user descriptor
  g0[1] = lds_addr;                             // LDS byte address
  g0[2] = (unsigned)gaddr;                      // global_addr[31:0]
  g0[3] = ((unsigned)(gaddr >> 32) & 0x01ffffffu) | (2u << 30);  // addr[56:32], type=2
  int32x4 gz = (int32x4)0;
#if defined(__clang_major__) && (__clang_major__ >= 23)
  int32x8 gz8 = (int32x8)0;
  __builtin_amdgcn_tensor_load_to_lds(g0, g1, gz, gz, gz8, 0);
#else
  __builtin_amdgcn_tensor_load_to_lds(g0, g1, gz, gz, 0);
#endif
}
__device__ __forceinline__ void wait_tensor_le2() {
#if __has_builtin(__builtin_amdgcn_s_wait_tensorcnt)
  __builtin_amdgcn_s_wait_tensorcnt(2);
#else
  asm volatile("s_wait_tensorcnt 0x2" ::: "memory");
#endif
}
__device__ __forceinline__ void wait_tensor_le0() {
#if __has_builtin(__builtin_amdgcn_s_wait_tensorcnt)
  __builtin_amdgcn_s_wait_tensorcnt(0);
#else
  asm volatile("s_wait_tensorcnt 0x0" ::: "memory");
#endif
}
#endif  // USE_TDM

// ---------------- pre-pass: K -> f16, V -> f16 transposed, mask -> log2-bias
__global__ __launch_bounds__(256)
void prepass(const float* __restrict__ K, const float* __restrict__ V,
             const float* __restrict__ Mask, _Float16* __restrict__ Kh,
             _Float16* __restrict__ VT, float* __restrict__ BiasL,
             int S, int n4) {
  int idx = blockIdx.x * 256 + threadIdx.x;     // one float4 of K/V per thread
  if (idx >= n4) return;
  int bs = idx >> 4;                            // b*S + s
  int dq = (idx & 15) * 4;
  int b  = bs / S;
  int s  = bs - b * S;

  F4 kv; kv.v = *(const f4v*)(K + (size_t)idx * 4);
  union { h4v v; unsigned w[2]; } kh;
  kh.w[0] = pk2(kv.f[0], kv.f[1]);
  kh.w[1] = pk2(kv.f[2], kv.f[3]);
  *(h4v*)(Kh + (size_t)idx * 4) = kh.v;

  F4 vv; vv.v = *(const f4v*)(V + (size_t)idx * 4);
  _Float16* vtb = VT + (size_t)b * 64 * S;
#pragma unroll
  for (int j = 0; j < 4; ++j) vtb[(size_t)(dq + j) * S + s] = (_Float16)vv.f[j];

  if (dq == 0) BiasL[bs] = (Mask[bs] - 1.0f) * BIGL2;
}

// ------- flash attention: Sᵀ = K·Qᵀ, Oᵀ = Vᵀ·Pᵀ.  2 waves x 32 queries. -------
__global__ __launch_bounds__(64)
void fa_fwd_cdna5(const float* __restrict__ Q, const _Float16* __restrict__ Kh,
                  const _Float16* __restrict__ VT, const float* __restrict__ BiasL,
                  float* __restrict__ O, int S) {
  __shared__ __align__(16) _Float16 ldsK[2][32 * 64];  // [buf][key][d]
  __shared__ __align__(16) _Float16 ldsV[2][64 * 32];  // [buf][d][key]

  const int tid  = threadIdx.x;
  const int wave = __builtin_amdgcn_readfirstlane(tid >> 5);  // scalar wave id
  const int lane = tid & 31;
  const int h    = lane >> 4;
  const int n    = lane & 15;

  const int b     = blockIdx.y;
  const int qbase = blockIdx.x * 64;
  const int qw    = qbase + wave * 32;       // wave's lowest query (2 tiles)

  const _Float16* Khb = Kh + ((size_t)b * S) * 64;
  const _Float16* VTb = VT + ((size_t)b * 64) * S;
  const float*    Bb  = BiasL + (size_t)b * S;

  // ---- Q fragments for 2 query tiles (B layout: elem e<8 -> d = 8h+e+32c, e>=8 -> d = 16+8h+(e-8)+32c)
  Frag16 qf[2][2];
  int qq[2];
#pragma unroll
  for (int u = 0; u < 2; ++u) {
    qq[u] = qw + 16 * u + n;
    const float* qrow = Q + (((size_t)b * S) + qq[u]) * 64;
#pragma unroll
    for (int c = 0; c < 2; ++c) {
      const float* p = qrow + 32 * c + 8 * h;
#pragma unroll
      for (int i = 0; i < 2; ++i) {
        F4 t0; t0.v = *(const f4v*)(p + 16 * i);
        F4 t1; t1.v = *(const f4v*)(p + 16 * i + 4);
        qf[u][c].w[4 * i + 0] = pk2(t0.f[0], t0.f[1]);
        qf[u][c].w[4 * i + 1] = pk2(t0.f[2], t0.f[3]);
        qf[u][c].w[4 * i + 2] = pk2(t1.f[0], t1.f[1]);
        qf[u][c].w[4 * i + 3] = pk2(t1.f[2], t1.f[3]);
      }
    }
  }

  v8f accO[2][4] = {};
  float m[2] = {-1e30f, -1e30f}, l[2] = {0.0f, 0.0f};
  const int kend = qbase + 64;

#if USE_TDM
  const int32x8 g1K = tdm_g1(64u, 32u, 64u, (unsigned)S, 64ull);
  const int32x8 g1V = tdm_g1(32u, 64u, (unsigned)S, 64u, (unsigned long long)S);
  const unsigned long long kAddr0 = (unsigned long long)(uintptr_t)Khb;
  const unsigned long long vAddr0 = (unsigned long long)(uintptr_t)VTb;
  unsigned ldsKa[2], ldsVa[2];
#pragma unroll
  for (int i = 0; i < 2; ++i) {
    ldsKa[i] = (unsigned)(uintptr_t)&ldsK[i][0];
    ldsVa[i] = (unsigned)(uintptr_t)&ldsV[i][0];
  }
  if (wave == 0) {
    tdm_issue(ldsKa[0], kAddr0, g1K);
    tdm_issue(ldsVa[0], vAddr0, g1V);
  }
#endif

  for (int kstep = 0; kstep < kend; kstep += 32) {
    const int cur = (kstep >> 5) & 1;
#if USE_TDM
    if (wave == 0) {
      if (kstep + 32 < kend) {
        tdm_issue(ldsKa[cur ^ 1], kAddr0 + (unsigned long long)(kstep + 32) * 128ull, g1K);
        tdm_issue(ldsVa[cur ^ 1], vAddr0 + (unsigned long long)(kstep + 32) * 2ull, g1V);
        wait_tensor_le2();
      } else {
        wait_tensor_le0();
      }
    }
    __syncthreads();                      // release staged tile to all waves
#else
    __syncthreads();
#pragma unroll
    for (int jj = 0; jj < 4; ++jj) {
      int i4 = tid + 64 * jj;             // 256 x 16B per tile
      *(u4v*)&ldsK[cur][i4 * 8] =
          *(const u4v*)(Khb + (size_t)kstep * 64 + i4 * 8);
      int d = i4 >> 2, kg = i4 & 3;
      *(u4v*)&ldsV[cur][d * 32 + kg * 8] =
          *(const u4v*)(VTb + (size_t)d * S + kstep + kg * 8);
    }
    __syncthreads();
#endif

    const bool act0 = (kstep <= qw + 15);       // tile 0 has unmasked keys
    const bool act1 = (kstep <= qw + 31);       // tile 1 has unmasked keys
    if (act1) {
      // ---- K fragments (A layout), shared by both query tiles
      Frag16 kf[2][2];                    // [key tile][d chunk]
#pragma unroll
      for (int c = 0; c < 2; ++c) {
        int off0 = n * 64 + 32 * c + 8 * h;
        int off1 = (16 + n) * 64 + 32 * c + 8 * h;
        kf[0][c].u4[0] = *(const u4v*)(&ldsK[cur][off0]);
        kf[0][c].u4[1] = *(const u4v*)(&ldsK[cur][off0 + 16]);
        kf[1][c].u4[0] = *(const u4v*)(&ldsK[cur][off1]);
        kf[1][c].u4[1] = *(const u4v*)(&ldsK[cur][off1 + 16]);
      }

      // ---- per-key padding bias (shared by both query tiles)
      F4 b0a, b0b, b1a, b1b;
      b0a.v = *(const f4v*)(Bb + kstep + 8 * h);
      b0b.v = *(const f4v*)(Bb + kstep + 8 * h + 4);
      b1a.v = *(const f4v*)(Bb + kstep + 16 + 8 * h);
      b1b.v = *(const f4v*)(Bb + kstep + 16 + 8 * h + 4);
      float blb[16];
#pragma unroll
      for (int r = 0; r < 4; ++r) {
        blb[r] = b0a.f[r]; blb[4 + r]  = b0b.f[r];
        blb[8 + r] = b1a.f[r]; blb[12 + r] = b1b.f[r];
      }

      Frag16 pB[2];
#pragma unroll
      for (int u = 0; u < 2; ++u) {
        const bool actU = u ? act1 : act0;
        if (actU) {
          // ---- Sᵀ = K · Qᵀ for query tile u
          v8f s0 = {}, s1 = {};
#pragma unroll
          for (int c = 0; c < 2; ++c) {
            s0 = __builtin_amdgcn_wmma_f32_16x16x32_f16(false, kf[0][c].v, false,
                                                        qf[u][c].v, (short)0, s0,
                                                        false, false);
            s1 = __builtin_amdgcn_wmma_f32_16x16x32_f16(false, kf[1][c].v, false,
                                                        qf[u][c].v, (short)0, s1,
                                                        false, false);
          }

          // ---- logits in log2 domain: tl = s*scale + padding bias
          float tl[16];
#pragma unroll
          for (int r = 0; r < 8; ++r) {
            tl[r]     = fmaf(s0[r], SCL2E, blb[r]);
            tl[8 + r] = fmaf(s1[r], SCL2E, blb[8 + r]);
          }
          // causal terms folded into tl only on the (rare) diagonal blocks
          if (kstep + 31 > qw + 16 * u) {
            int dd = qq[u] - kstep - 8 * h;     // key index r masked iff r > dd
#pragma unroll
            for (int r = 0; r < 8; ++r) {
              tl[r]     += (r > dd)      ? -BIGL2 : 0.0f;
              tl[8 + r] += (r + 16 > dd) ? -BIGL2 : 0.0f;
            }
          }

          // ---- online softmax
          float ms = tl[0];
#pragma unroll
          for (int i = 1; i < 16; ++i) ms = fmaxf(ms, tl[i]);
          ms = fmaxf(ms, __shfl_xor(ms, 16));
          float mn    = fmaxf(m[u], ms);
          float alpha = __builtin_amdgcn_exp2f(m[u] - mn);
          float p[16]; float ps = 0.0f;
#pragma unroll
          for (int i = 0; i < 16; ++i) {
            p[i] = __builtin_amdgcn_exp2f(tl[i] - mn);
            ps += p[i];
          }
          ps += __shfl_xor(ps, 16);
          l[u] = l[u] * alpha + ps;
          m[u] = mn;
#pragma unroll
          for (int t = 0; t < 4; ++t) accO[u][t] *= alpha;

          // ---- Pᵀ -> B layout (e<8 -> key = 8h+e, e>=8 -> key = 16+8h+(e-8))
          unsigned pw0[4], pw1[4];
#pragma unroll
          for (int i = 0; i < 4; ++i) {
            pw0[i] = pk2(p[2 * i], p[2 * i + 1]);      // keys  8h + 0..7
            pw1[i] = pk2(p[8 + 2 * i], p[9 + 2 * i]);  // keys  16+8h + 0..7
          }
          unsigned x0[4], x1[4];
#pragma unroll
          for (int i = 0; i < 4; ++i) {
            x0[i] = __shfl_xor(pw0[i], 16);
            x1[i] = __shfl_xor(pw1[i], 16);
          }
#pragma unroll
          for (int i = 0; i < 4; ++i) {
            pB[u].w[i]     = pw0[i];                   // keys  8h + 0..7
            pB[u].w[4 + i] = pw1[i];                   // keys  16+8h + 0..7
          }
        }
      }

      // ---- Oᵀ += Vᵀ · Pᵀ : V fragments shared by both query tiles
#pragma unroll
      for (int t = 0; t < 4; ++t) {
        Frag16 vf;                         // A layout: lane = d row (16t+n)
        int off = (16 * t + n) * 32 + 8 * h;
        vf.u4[0] = *(const u4v*)(&ldsV[cur][off]);
        vf.u4[1] = *(const u4v*)(&ldsV[cur][off + 16]);
        if (act0)
          accO[0][t] = __builtin_amdgcn_wmma_f32_16x16x32_f16(
              false, vf.v, false, pB[0].v, (short)0, accO[0][t], false, false);
        accO[1][t] = __builtin_amdgcn_wmma_f32_16x16x32_f16(
            false, vf.v, false, pB[1].v, (short)0, accO[1][t], false, false);
      }
    }  // act1
#if USE_TDM
    __syncthreads();                      // all reads done before buf reuse
#endif
  }  // kstep

  // ---- epilogue: normalize and store both query tiles
#pragma unroll
  for (int u = 0; u < 2; ++u) {
    float inv = 1.0f / fmaxf(l[u], 1e-30f);
    float* orow = O + ((size_t)b * S + qq[u]) * 64;
#pragma unroll
    for (int t = 0; t < 4; ++t) {
      F4 o0, o1;
#pragma unroll
      for (int r = 0; r < 4; ++r) {
        o0.f[r] = accO[u][t][r] * inv;
        o1.f[r] = accO[u][t][4 + r] * inv;
      }
      *(f4v*)(orow + 16 * t + 8 * h)     = o0.v;
      *(f4v*)(orow + 16 * t + 8 * h + 4) = o1.v;
    }
  }
}

extern "C" void kernel_launch(void* const* d_in, const int* in_sizes, int n_in,
                              void* d_out, int out_size, void* d_ws, size_t ws_size,
                              hipStream_t stream) {
  const float* Qp = (const float*)d_in[0];
  const float* Kp = (const float*)d_in[1];
  const float* Vp = (const float*)d_in[2];
  const float* Mp = (const float*)d_in[3];
  float* Op = (float*)d_out;

  const int S = 4096;                 // reference: S=4096, D=64
  const int B = in_sizes[3] / S;      // mask is [B, S]

  // workspace layout: Kh f16 | VT f16 | BiasL f32
  _Float16* Kh = (_Float16*)d_ws;
  _Float16* VT = Kh + (size_t)B * S * 64;
  float* BiasL = (float*)(VT + (size_t)B * S * 64);

  int n4 = B * S * 16;                // float4s of K/V
  prepass<<<dim3((n4 + 255) / 256), dim3(256), 0, stream>>>(
      Kp, Vp, Mp, Kh, VT, BiasL, S, n4);

  fa_fwd_cdna5<<<dim3(S / 64, B), dim3(64), 0, stream>>>(
      Qp, Kh, VT, BiasL, Op, S);
}
